// ShapeDeformCardiac3DCircleNet_47966194762186
// MI455X (gfx1250) — matrix operand order repaired and checked by reading, not compiled, not run on hardware
//
#include <hip/hip_runtime.h>
#include <math.h>

typedef __attribute__((ext_vector_type(16))) __bf16 v16bf;
typedef __attribute__((ext_vector_type(8)))  float  v8f;

union BF16x16 { v16bf v; unsigned short u[16]; };

__device__ inline unsigned short f2bf(float f){
  unsigned u = __builtin_bit_cast(unsigned, f);
  u += 0x7FFFu + ((u >> 16) & 1u);            // round-to-nearest-even
  return (unsigned short)(u >> 16);
}
__device__ inline float bf2f(unsigned short h){
  return __builtin_bit_cast(float, (unsigned)h << 16);
}

// ---------------- input normalization ----------------
__global__ void k_minmax(const float* __restrict__ img, float* __restrict__ mm){
  int b = blockIdx.x, t = threadIdx.x;
  const float* p = img + (size_t)b * 262144;
  float lo = 1e30f, hi = -1e30f;
  for (int i = t; i < 262144; i += 256){ float v = p[i]; lo = fminf(lo, v); hi = fmaxf(hi, v); }
  __shared__ float slo[256], shi[256];
  slo[t] = lo; shi[t] = hi; __syncthreads();
  for (int s = 128; s > 0; s >>= 1){
    if (t < s){ slo[t] = fminf(slo[t], slo[t+s]); shi[t] = fmaxf(shi[t], shi[t+s]); }
    __syncthreads();
  }
  if (!t){ mm[b*2] = slo[0]; mm[b*2+1] = shi[0]; }
}

__global__ void k_norm(const float* __restrict__ img, const float* __restrict__ mm,
                       unsigned short* __restrict__ x0){
  int i = blockIdx.x*256 + threadIdx.x;
  if (i >= 1048576) return;
  int b = i >> 18;
  float mn = mm[b*2], mx = mm[b*2+1];
  x0[i] = f2bf((img[i] - mn) / (mx - mn + 0.01f));
}

// ---------------- weight repack to bf16 [K][N] ----------------
__global__ void k_wconv(const float* __restrict__ w, unsigned short* __restrict__ wb,
                        int Cin, int N, int Ktaps){
  int K = Ktaps * Cin;
  int idx = blockIdx.x*256 + threadIdx.x;
  if (idx >= K*N) return;
  int k = idx / N, n = idx - k*N;
  int tap = k / Cin, cin = k - tap*Cin;
  wb[idx] = f2bf(w[((size_t)n*Cin + cin)*Ktaps + tap]);
}

// fc1: our channels-last flatten k = s*128+c maps to reference NCDHW flatten c*8+s
__global__ void k_wfc1(const float* __restrict__ w, unsigned short* __restrict__ wb){
  int idx = blockIdx.x*256 + threadIdx.x;
  if (idx >= 1024*400) return;
  int k = idx / 400, n = idx - k*400;
  int s = k >> 7, c = k & 127;
  wb[idx] = f2bf(w[n*1024 + c*8 + s]);
}

// ---------------- implicit-GEMM conv via WMMA bf16 ----------------
// Block tile 64(M) x 64(N); 4 waves; each wave: 16x64 via 4 accumulators,
// reusing one A fragment across 4 WMMAs per K-step.
// act: [B][Di][Di][Di][Cin] bf16 (channels-last), wmat: [K][N] bf16, z: [M][N] f32
__global__ __launch_bounds__(128)
void k_conv_wmma(const unsigned short* __restrict__ act,
                 const unsigned short* __restrict__ wmat,
                 const float* __restrict__ bias,
                 float* __restrict__ z,
                 int Bn, int Di, int Do, int Cin, int Cout,
                 int stride, int pad, int Ktaps)
{
  const int K   = Ktaps * Cin;
  const int Dov = Do*Do*Do;
  const int M   = Bn * Dov;
  const int N   = Cout;
  __shared__ unsigned short As[64][32];   // [m][k]
  __shared__ unsigned short Bs[64][32];   // [n][k] (transposed for contiguous frag reads)
  const int tid  = threadIdx.x;
  const int wv   = tid >> 5;
  const int ln   = tid & 31;
  const int half = ln >> 4;
  const int l16  = ln & 15;
  const int m0   = blockIdx.x * 64;
  const int n0   = blockIdx.y * 64;
  v8f acc0 = {0.f,0.f,0.f,0.f,0.f,0.f,0.f,0.f};
  v8f acc1 = acc0, acc2 = acc0, acc3 = acc0;

  for (int k0 = 0; k0 < K; k0 += 32){
    __syncthreads();
    // A tile: 64 rows x 32 k (implicit im2col gather)
    for (int idx = tid; idx < 64*32; idx += 128){
      int r = idx >> 5, c = idx & 31;
      int m = m0 + r, kk = k0 + c;
      unsigned short bits = 0;
      if (m < M && kk < K){
        int tap = kk / Cin, cin = kk - tap*Cin;
        int kz = tap/25, kt = tap - kz*25, ky = kt/5, kx = kt - ky*5;
        int b  = m / Dov, sp = m - b*Dov;
        int oz = sp/(Do*Do), s2 = sp - oz*Do*Do, oy = s2/Do, ox = s2 - oy*Do;
        int iz = oz*stride - pad + kz;
        int iy = oy*stride - pad + ky;
        int ix = ox*stride - pad + kx;
        if ((unsigned)iz < (unsigned)Di && (unsigned)iy < (unsigned)Di && (unsigned)ix < (unsigned)Di)
          bits = act[((((size_t)b*Di + iz)*Di + iy)*Di + ix)*Cin + cin];
      }
      As[r][c] = bits;
    }
    // B tile: 32 k x 64 n  ->  Bs[n][k]
    for (int idx = tid; idx < 64*32; idx += 128){
      int n = idx & 63, kk = idx >> 6;
      unsigned short bits = 0;
      if (k0 + kk < K && n0 + n < N) bits = wmat[(size_t)(k0+kk)*N + (n0+n)];
      Bs[n][kk] = bits;
    }
    __syncthreads();
    if (k0 + 32 < K) __builtin_prefetch(&wmat[(size_t)(k0+32)*N + n0], 0, 1);
    // wave32 fragments: contiguous k = half*16 + j in both A and B
    BF16x16 af, bf0, bf1, bf2, bf3;
    #pragma unroll
    for (int j = 0; j < 16; ++j){
      af.u[j]   = As[wv*16 + l16][half*16 + j];
    }
    #pragma unroll
    for (int j = 0; j < 16; ++j){
      bf0.u[j] = Bs[     l16][half*16 + j];
      bf1.u[j] = Bs[16 + l16][half*16 + j];
      bf2.u[j] = Bs[32 + l16][half*16 + j];
      bf3.u[j] = Bs[48 + l16][half*16 + j];
    }
    acc0 = __builtin_amdgcn_wmma_f32_16x16x32_bf16(false, af.v, false, bf0.v, (short)0, acc0, false, false);
    acc1 = __builtin_amdgcn_wmma_f32_16x16x32_bf16(false, af.v, false, bf1.v, (short)0, acc1, false, false);
    acc2 = __builtin_amdgcn_wmma_f32_16x16x32_bf16(false, af.v, false, bf2.v, (short)0, acc2, false, false);
    acc3 = __builtin_amdgcn_wmma_f32_16x16x32_bf16(false, af.v, false, bf3.v, (short)0, acc3, false, false);
  }
  // C layout (ISA 7.12.2): VGPR r -> row r + 8*half, col = lane%16
  int row = m0 + wv*16 + half*8;
  #pragma unroll
  for (int r = 0; r < 8; ++r){
    if (row + r < M){
      int c0c = n0 + l16;
      if (c0c      < N) z[(size_t)(row+r)*N + c0c     ] = acc0[r] + bias[c0c];
      if (c0c + 16 < N) z[(size_t)(row+r)*N + c0c + 16] = acc1[r] + bias[c0c+16];
      if (c0c + 32 < N) z[(size_t)(row+r)*N + c0c + 32] = acc2[r] + bias[c0c+32];
      if (c0c + 48 < N) z[(size_t)(row+r)*N + c0c + 48] = acc3[r] + bias[c0c+48];
    }
  }
}

// ---------------- dense GEMM via WMMA (FC layers), same 64x64 tiling ----------------
__global__ __launch_bounds__(128)
void k_gemm_wmma(const unsigned short* __restrict__ Am,
                 const unsigned short* __restrict__ Bm,
                 const float* __restrict__ bias,
                 float* __restrict__ z,
                 int M, int K, int N)
{
  __shared__ unsigned short As[64][32];
  __shared__ unsigned short Bs[64][32];
  const int tid  = threadIdx.x;
  const int wv   = tid >> 5;
  const int ln   = tid & 31;
  const int half = ln >> 4;
  const int l16  = ln & 15;
  const int m0   = blockIdx.x * 64;
  const int n0   = blockIdx.y * 64;
  v8f acc0 = {0.f,0.f,0.f,0.f,0.f,0.f,0.f,0.f};
  v8f acc1 = acc0, acc2 = acc0, acc3 = acc0;

  for (int k0 = 0; k0 < K; k0 += 32){
    __syncthreads();
    for (int idx = tid; idx < 64*32; idx += 128){
      int r = idx >> 5, c = idx & 31;
      int m = m0 + r, kk = k0 + c;
      As[r][c] = (m < M && kk < K) ? Am[(size_t)m*K + kk] : (unsigned short)0;
    }
    for (int idx = tid; idx < 64*32; idx += 128){
      int n = idx & 63, kk = idx >> 6;
      Bs[n][kk] = (k0 + kk < K && n0 + n < N) ? Bm[(size_t)(k0+kk)*N + (n0+n)]
                                              : (unsigned short)0;
    }
    __syncthreads();
    BF16x16 af, bf0, bf1, bf2, bf3;
    #pragma unroll
    for (int j = 0; j < 16; ++j){
      af.u[j]   = As[wv*16 + l16][half*16 + j];
    }
    #pragma unroll
    for (int j = 0; j < 16; ++j){
      bf0.u[j] = Bs[     l16][half*16 + j];
      bf1.u[j] = Bs[16 + l16][half*16 + j];
      bf2.u[j] = Bs[32 + l16][half*16 + j];
      bf3.u[j] = Bs[48 + l16][half*16 + j];
    }
    acc0 = __builtin_amdgcn_wmma_f32_16x16x32_bf16(false, af.v, false, bf0.v, (short)0, acc0, false, false);
    acc1 = __builtin_amdgcn_wmma_f32_16x16x32_bf16(false, af.v, false, bf1.v, (short)0, acc1, false, false);
    acc2 = __builtin_amdgcn_wmma_f32_16x16x32_bf16(false, af.v, false, bf2.v, (short)0, acc2, false, false);
    acc3 = __builtin_amdgcn_wmma_f32_16x16x32_bf16(false, af.v, false, bf3.v, (short)0, acc3, false, false);
  }
  int row = m0 + wv*16 + half*8;
  #pragma unroll
  for (int r = 0; r < 8; ++r){
    if (row + r < M){
      int c0c = n0 + l16;
      if (c0c      < N) z[(size_t)(row+r)*N + c0c     ] = acc0[r] + bias[c0c];
      if (c0c + 16 < N) z[(size_t)(row+r)*N + c0c + 16] = acc1[r] + bias[c0c+16];
      if (c0c + 32 < N) z[(size_t)(row+r)*N + c0c + 32] = acc2[r] + bias[c0c+32];
      if (c0c + 48 < N) z[(size_t)(row+r)*N + c0c + 48] = acc3[r] + bias[c0c+48];
    }
  }
}

// ---------------- batch-norm ----------------
__global__ void k_bnstats(const float* __restrict__ zbuf, float* __restrict__ mean,
                          float* __restrict__ var, int M, int C){
  int c = blockIdx.x, t = threadIdx.x;
  float s = 0.f, q = 0.f;
  for (int m = t; m < M; m += 256){ float v = zbuf[(size_t)m*C + c]; s += v; q += v*v; }
  __shared__ float ss[256], qq[256];
  ss[t] = s; qq[t] = q; __syncthreads();
  for (int st = 128; st > 0; st >>= 1){
    if (t < st){ ss[t] += ss[t+st]; qq[t] += qq[t+st]; }
    __syncthreads();
  }
  if (!t){ float mu = ss[0]/(float)M; mean[c] = mu; var[c] = qq[0]/(float)M - mu*mu; }
}

__global__ void k_bnrelu(const float* __restrict__ zbuf, const float* __restrict__ mean,
                         const float* __restrict__ var, const float* __restrict__ g,
                         const float* __restrict__ bta, unsigned short* __restrict__ out,
                         int M, int C){
  int i = blockIdx.x*256 + threadIdx.x;
  if (i >= M*C) return;
  int c = i % C;
  float v = (zbuf[i] - mean[c]) * rsqrtf(var[c] + 1e-5f) * g[c] + bta[c];
  out[i] = f2bf(fmaxf(v, 0.f));
}

// ---------------- 2x2x2 max-pool (4^3 -> 2^3, 256 ch, channels-last) ----------------
__global__ void k_maxpool(const unsigned short* __restrict__ a, unsigned short* __restrict__ o){
  int i = blockIdx.x*256 + threadIdx.x;
  if (i >= 32*256) return;
  int c = i & 255, m = i >> 8;
  int b = m >> 3, sp = m & 7;
  int pz = sp >> 2, py = (sp >> 1) & 1, px = sp & 1;
  float best = -1e30f;
  for (int dz = 0; dz < 2; ++dz)
    for (int dy = 0; dy < 2; ++dy)
      for (int dx = 0; dx < 2; ++dx)
        best = fmaxf(best, bf2f(a[(((b*4 + pz*2+dz)*4 + py*2+dy)*4 + px*2+dx)*256 + c]));
  o[i] = f2bf(best);
}

// ---------------- regression heads ----------------
__global__ void k_heads(const unsigned short* __restrict__ af2,
                        const float* __restrict__ e1w, const float* __restrict__ e1b,
                        const float* __restrict__ e2w, const float* __restrict__ e2b,
                        float* __restrict__ par1, float* __restrict__ par2){
  int t = threadIdx.x;
  if (t >= 16) return;
  int b = t >> 2, o = t & 3;
  float d = 0.f;
  if (o < 3){
    for (int k = 0; k < 200; ++k) d += bf2f(af2[b*200+k]) * e1w[o*200+k];
    par1[b*3+o] = tanhf(d + e1b[o]);
  } else {
    for (int k = 0; k < 200; ++k) d += bf2f(af2[b*200+k]) * e2w[k];
    par2[b] = 1.f / (1.f + __expf(-(d + e2b[0])));
  }
}

// ---------------- tube nodes ----------------
__global__ void k_nodes(const float* __restrict__ par1, const float* __restrict__ par2,
                        float* __restrict__ nodes){
  int id = blockIdx.x*128 + threadIdx.x;
  if (id >= 4*132) return;
  int b = id / 132, nid = id - b*132;
  int s = nid / 33, j = nid - s*33;
  float cx = par1[b*3+0]*0.3f, cy = par1[b*3+1]*0.3f, cz = par1[b*3+2]*0.2f;
  float r  = par2[b]*0.5f + 0.05f;
  float zz = -0.6f + 0.4f*(float)s + cz;
  float x = cx, y = cy;
  if (j > 0){
    float th = (float)(j-1) * (2.f*3.14159265358979323846f/32.f);
    x = cx + r*cosf(th); y = cy + r*sinf(th);
  }
  float* nd = nodes + (size_t)(b*132 + nid)*3;
  nd[0] = x; nd[1] = y; nd[2] = zz;
}

// ---------------- tetra face planes (288 tets x 4 faces per batch) ----------------
__global__ void k_planes(const float* __restrict__ nodes, float* __restrict__ planes){
  int id = blockIdx.x*128 + threadIdx.x;
  if (id >= 4*288) return;
  int b = id / 288, t = id - b*288;
  int g = t / 96, rr = t - g*96, j = rr/3, w = rr - j*3;
  int c0 = g*33, c1 = c0 + 33;
  int jn = (j + 1) & 31;
  int p[6] = { c0, c0+1+j, c0+1+jn, c1, c1+1+j, c1+1+jn };
  int vids[4] = { p[w], p[w+1], p[w+2], p[w+3] };
  float vx[4], vy[4], vz[4];
  for (int i = 0; i < 4; ++i){
    const float* nd = nodes + ((size_t)b*132 + vids[i])*3;
    vx[i] = nd[0]; vy[i] = nd[1]; vz[i] = nd[2];
  }
  const int f0[4] = {1,0,0,0}, f1[4] = {2,3,1,2}, f2[4] = {3,2,3,1};
  for (int f = 0; f < 4; ++f){
    int i0 = f0[f], i1 = f1[f], i2 = f2[f];
    float e1x = vx[i1]-vx[i0], e1y = vy[i1]-vy[i0], e1z = vz[i1]-vz[i0];
    float e2x = vx[i2]-vx[i0], e2y = vy[i2]-vy[i0], e2z = vz[i2]-vz[i0];
    float nx = e1y*e2z - e1z*e2y;
    float ny = e1z*e2x - e1x*e2z;
    float nz = e1x*e2y - e1y*e2x;
    float sgd = nx*(vx[f]-vx[i0]) + ny*(vy[f]-vy[i0]) + nz*(vz[f]-vz[i0]);
    float sgn = (sgd > 0.f) ? 1.f : ((sgd < 0.f) ? -1.f : 0.f);
    nx *= sgn; ny *= sgn; nz *= sgn;
    float inv = 1.f / (sqrtf(nx*nx + ny*ny + nz*nz) + 1e-8f);
    nx *= inv; ny *= inv; nz *= inv;
    float off = -(nx*vx[i0] + ny*vy[i0] + nz*vz[i0]);
    float* pl = planes + (((size_t)b*288 + t)*4 + f)*4;
    pl[0] = nx; pl[1] = ny; pl[2] = nz; pl[3] = off;
  }
}

// ---------------- voxelize: async-LDS plane staging + FMA sweep ----------------
// pl[] is the ONLY LDS object in this kernel -> allocated at LDS offset 0, so
// lane-local LDS byte addresses for the async copy are simply tid*16.
__global__ __launch_bounds__(256)
void k_voxelize(const float* __restrict__ planes, float* __restrict__ out){
  __shared__ float pl[48*16];
  int b = blockIdx.y;
  int L = blockIdx.x*256 + threadIdx.x;
  int iz = L >> 12, iy = (L >> 6) & 63, ix = L & 63;
  float px = -1.f + (2.f/63.f)*(float)ix;
  float py = -1.f + (2.f/63.f)*(float)iy;
  float pz = -1.f + (2.f/63.f)*(float)iz;
  float occ = -1e30f;
  for (int t0 = 0; t0 < 288; t0 += 48){
    __syncthreads();                       // previous chunk fully consumed
    if (threadIdx.x < 192){                // 192 lanes x 16B = 48 tets * 4 planes * 16B
      unsigned loff = threadIdx.x * 16u;                       // LDS byte addr (base 0)
      unsigned gofs = (unsigned)((b*288 + t0) * 64) + threadIdx.x * 16u; // byte offset
      asm volatile("global_load_async_to_lds_b128 %0, %1, %2"
                   :: "v"(loff), "v"(gofs), "s"(planes) : "memory");
    }
    asm volatile("s_wait_asynccnt 0x0" ::: "memory");
    __syncthreads();                       // all waves' async copies visible
    #pragma unroll 4
    for (int t = 0; t < 48; ++t){
      const float* q = &pl[t*16];
      float d0 = fmaf(q[0], px, fmaf(q[1], py, fmaf(q[2], pz, q[3])));
      float d1 = fmaf(q[4], px, fmaf(q[5], py, fmaf(q[6], pz, q[7])));
      float d2 = fmaf(q[8], px, fmaf(q[9], py, fmaf(q[10], pz, q[11])));
      float d3 = fmaf(q[12], px, fmaf(q[13], py, fmaf(q[14], pz, q[15])));
      occ = fmaxf(occ, fminf(fminf(d0, d1), fminf(d2, d3)));
    }
  }
  out[(size_t)b*262144 + L] = 1.f / (1.f + __expf(-occ*50.f));
}

// ---------------- host launch ----------------
extern "C" void kernel_launch(void* const* d_in, const int* in_sizes, int n_in,
                              void* d_out, int out_size, void* d_ws, size_t ws_size,
                              hipStream_t stream)
{
  (void)in_sizes; (void)n_in; (void)out_size; (void)ws_size;
  const float* img = (const float*)d_in[0];
  const float* ew[4]  = {(const float*)d_in[1],(const float*)d_in[5],(const float*)d_in[9],(const float*)d_in[13]};
  const float* eb[4]  = {(const float*)d_in[2],(const float*)d_in[6],(const float*)d_in[10],(const float*)d_in[14]};
  const float* eg[4]  = {(const float*)d_in[3],(const float*)d_in[7],(const float*)d_in[11],(const float*)d_in[15]};
  const float* ebb[4] = {(const float*)d_in[4],(const float*)d_in[8],(const float*)d_in[12],(const float*)d_in[16]};
  const float* rw0=(const float*)d_in[17]; const float* rb0=(const float*)d_in[18];
  const float* rg0=(const float*)d_in[19]; const float* rbb0=(const float*)d_in[20];
  const float* rw1=(const float*)d_in[21]; const float* rb1=(const float*)d_in[22];
  const float* rg1=(const float*)d_in[23]; const float* rbb1=(const float*)d_in[24];
  const float* f1w=(const float*)d_in[25]; const float* f1b=(const float*)d_in[26];
  const float* f1g=(const float*)d_in[27]; const float* f1bb=(const float*)d_in[28];
  const float* f2w=(const float*)d_in[29]; const float* f2b=(const float*)d_in[30];
  const float* f2g=(const float*)d_in[31]; const float* f2bb=(const float*)d_in[32];
  const float* e1w=(const float*)d_in[33]; const float* e1b=(const float*)d_in[34];
  const float* e2w=(const float*)d_in[35]; const float* e2b=(const float*)d_in[36];
  float* out = (float*)d_out;

  char* ws = (char*)d_ws;
  size_t off = 0;
  auto AL = [&](size_t bytes)->void*{ off = (off + 255) & ~(size_t)255; void* p = ws + off; off += bytes; return p; };

  unsigned short* X0  = (unsigned short*)AL(1048576ull*2);
  unsigned short* WB0 = (unsigned short*)AL(8000ull*2);
  unsigned short* WB1 = (unsigned short*)AL(1024000ull*2);
  unsigned short* WB2 = (unsigned short*)AL(4096000ull*2);
  unsigned short* WB3 = (unsigned short*)AL(16384000ull*2);
  unsigned short* WR0 = (unsigned short*)AL(16384000ull*2);
  unsigned short* WR1 = (unsigned short*)AL(32768ull*2);
  unsigned short* WF1 = (unsigned short*)AL(409600ull*2);
  unsigned short* WF2 = (unsigned short*)AL(80000ull*2);
  float* Z1  = (float*)AL(8388608ull*4);
  float* Z2  = (float*)AL(2097152ull*4);
  float* Z3  = (float*)AL(524288ull*4);
  float* Z4  = (float*)AL(131072ull*4);
  float* ZR0 = (float*)AL(65536ull*4);
  float* ZR1 = (float*)AL(4096ull*4);
  float* ZF1 = (float*)AL(1600ull*4);
  float* ZF2 = (float*)AL(800ull*4);
  unsigned short* A1  = (unsigned short*)AL(8388608ull*2);
  unsigned short* A2  = (unsigned short*)AL(2097152ull*2);
  unsigned short* A3  = (unsigned short*)AL(524288ull*2);
  unsigned short* A4  = (unsigned short*)AL(131072ull*2);
  unsigned short* AR0 = (unsigned short*)AL(65536ull*2);
  unsigned short* PO  = (unsigned short*)AL(8192ull*2);
  unsigned short* AR1 = (unsigned short*)AL(4096ull*2);
  unsigned short* AF1 = (unsigned short*)AL(1600ull*2);
  unsigned short* AF2 = (unsigned short*)AL(800ull*2);
  float* MEAN = (float*)AL(512ull*4);
  float* VAR  = (float*)AL(512ull*4);
  float* MM   = (float*)AL(8ull*4);
  float* PAR1 = (float*)AL(12ull*4);
  float* PAR2 = (float*)AL(4ull*4);
  float* NODES= (float*)AL(1584ull*4);
  float* PLN  = (float*)AL(18432ull*4);

  auto cdiv = [](int a, int b){ return (a + b - 1) / b; };

  k_minmax<<<4, 256, 0, stream>>>(img, MM);
  k_norm<<<cdiv(1048576,256), 256, 0, stream>>>(img, MM, X0);

  k_wconv<<<cdiv(8000,256),     256, 0, stream>>>(ew[0], WB0,   1,  64, 125);
  k_wconv<<<cdiv(1024000,256),  256, 0, stream>>>(ew[1], WB1,  64, 128, 125);
  k_wconv<<<cdiv(4096000,256),  256, 0, stream>>>(ew[2], WB2, 128, 256, 125);
  k_wconv<<<cdiv(16384000,256), 256, 0, stream>>>(ew[3], WB3, 256, 512, 125);
  k_wconv<<<cdiv(16384000,256), 256, 0, stream>>>(rw0,   WR0, 512, 256, 125);
  k_wconv<<<cdiv(32768,256),    256, 0, stream>>>(rw1,   WR1, 256, 128, 1);
  k_wfc1 <<<cdiv(409600,256),   256, 0, stream>>>(f1w,   WF1);
  k_wconv<<<cdiv(80000,256),    256, 0, stream>>>(f2w,   WF2, 400, 200, 1);

  // encoder (grid.y = ceil(N/64))
  k_conv_wmma<<<dim3(cdiv(131072,64), 1), 128, 0, stream>>>(X0, WB0, eb[0], Z1, 4,64,32,  1, 64, 2,2,125);
  k_bnstats<<<64, 256, 0, stream>>>(Z1, MEAN, VAR, 131072, 64);
  k_bnrelu<<<cdiv(8388608,256), 256, 0, stream>>>(Z1, MEAN, VAR, eg[0], ebb[0], A1, 131072, 64);

  k_conv_wmma<<<dim3(cdiv(16384,64), 2), 128, 0, stream>>>(A1, WB1, eb[1], Z2, 4,32,16, 64,128, 2,2,125);
  k_bnstats<<<128, 256, 0, stream>>>(Z2, MEAN, VAR, 16384, 128);
  k_bnrelu<<<cdiv(2097152,256), 256, 0, stream>>>(Z2, MEAN, VAR, eg[1], ebb[1], A2, 16384, 128);

  k_conv_wmma<<<dim3(cdiv(2048,64), 4), 128, 0, stream>>>(A2, WB2, eb[2], Z3, 4,16, 8,128,256, 2,2,125);
  k_bnstats<<<256, 256, 0, stream>>>(Z3, MEAN, VAR, 2048, 256);
  k_bnrelu<<<cdiv(524288,256), 256, 0, stream>>>(Z3, MEAN, VAR, eg[2], ebb[2], A3, 2048, 256);

  k_conv_wmma<<<dim3(cdiv(256,64), 8), 128, 0, stream>>>(A3, WB3, eb[3], Z4, 4, 8, 4,256,512, 2,2,125);
  k_bnstats<<<512, 256, 0, stream>>>(Z4, MEAN, VAR, 256, 512);
  k_bnrelu<<<cdiv(131072,256), 256, 0, stream>>>(Z4, MEAN, VAR, eg[3], ebb[3], A4, 256, 512);

  // regression trunk
  k_conv_wmma<<<dim3(4, 4), 128, 0, stream>>>(A4, WR0, rb0, ZR0, 4, 4, 4, 512, 256, 1, 2, 125);
  k_bnstats<<<256, 256, 0, stream>>>(ZR0, MEAN, VAR, 256, 256);
  k_bnrelu<<<cdiv(65536,256), 256, 0, stream>>>(ZR0, MEAN, VAR, rg0, rbb0, AR0, 256, 256);

  k_maxpool<<<32, 256, 0, stream>>>(AR0, PO);

  k_conv_wmma<<<dim3(1, 2), 128, 0, stream>>>(PO, WR1, rb1, ZR1, 4, 2, 2, 256, 128, 1, 0, 1);
  k_bnstats<<<128, 256, 0, stream>>>(ZR1, MEAN, VAR, 32, 128);
  k_bnrelu<<<cdiv(4096,256), 256, 0, stream>>>(ZR1, MEAN, VAR, rg1, rbb1, AR1, 32, 128);

  k_gemm_wmma<<<dim3(1, 7), 128, 0, stream>>>(AR1, WF1, f1b, ZF1, 4, 1024, 400);
  k_bnstats<<<400, 256, 0, stream>>>(ZF1, MEAN, VAR, 4, 400);
  k_bnrelu<<<cdiv(1600,256), 256, 0, stream>>>(ZF1, MEAN, VAR, f1g, f1bb, AF1, 4, 400);

  k_gemm_wmma<<<dim3(1, 4), 128, 0, stream>>>(AF1, WF2, f2b, ZF2, 4, 400, 200);
  k_bnstats<<<200, 256, 0, stream>>>(ZF2, MEAN, VAR, 4, 200);
  k_bnrelu<<<cdiv(800,256), 256, 0, stream>>>(ZF2, MEAN, VAR, f2g, f2bb, AF2, 4, 200);

  // geometric decoder
  k_heads<<<1, 32, 0, stream>>>(AF2, e1w, e1b, e2w, e2b, PAR1, PAR2);
  k_nodes<<<cdiv(528,128), 128, 0, stream>>>(PAR1, PAR2, NODES);
  k_planes<<<cdiv(1152,128), 128, 0, stream>>>(NODES, PLN);
  k_voxelize<<<dim3(1024, 4), 256, 0, stream>>>(PLN, out);
}
